// MultiheadSelfAttention_88295937671356
// MI455X (gfx1250) — matrix operand order repaired and checked by reading, not compiled, not run on hardware
//
#include <hip/hip_runtime.h>
#include <hip/hip_bf16.h>
#include <stdint.h>

typedef __attribute__((ext_vector_type(16))) __bf16 v16bf;
typedef __attribute__((ext_vector_type(8)))  __bf16 v8bf;
typedef __attribute__((ext_vector_type(8)))  float  v8f;
typedef unsigned int uint32x4 __attribute__((ext_vector_type(4)));
typedef int          int32x4  __attribute__((ext_vector_type(4)));
typedef int          int32x8  __attribute__((ext_vector_type(8)));

#define D_MODEL   1024
#define NUM_HEADS 16
#define D_K       64
#define SEQ       2048
#define BATCH     2
#define M_TOTAL   (BATCH * SEQ)   // 4096

union BF16Frag {
  v16bf v;
  v8bf  h[2];
  unsigned short u[16];
  unsigned int   w[8];
};

__device__ __forceinline__ unsigned short f2bf(float f) {
  union { float f; unsigned u; } x; x.f = f;
  unsigned r = x.u + 0x7FFFu + ((x.u >> 16) & 1u);   // round-to-nearest-even
  return (unsigned short)(r >> 16);
}

__device__ __forceinline__ v8f wmma_bf16(v16bf a, v16bf b, v8f c) {
  // emits v_wmma_f32_16x16x32_bf16
  return __builtin_amdgcn_wmma_f32_16x16x32_bf16(false, a, false, b, (short)0, c,
                                                 false, false);
}

// ---------------------------------------------------------------------------
// TDM: DMA one W tile (tile_dim0=128 K-elems x tile_dim1=64 rows, bf16) from a
// row-major [1024 x 1024] bf16 tensor into LDS. D# packed per ISA 8.3/8.4.
// ---------------------------------------------------------------------------
__device__ __forceinline__ void tdm_load_w_tile(unsigned lds_byte, const void* gaddr) {
  const unsigned long long ga = (unsigned long long)(size_t)gaddr;
  uint32x4 g0;
  g0[0] = 1u;                                    // count=1 (valid user descriptor)
  g0[1] = lds_byte;                              // lds_addr [63:32]
  g0[2] = (unsigned)(ga & 0xFFFFFFFFu);          // global_addr [95:64]
  g0[3] = (unsigned)((ga >> 32) & 0x01FFFFFFu)   // global_addr [120:96]
        | 0x80000000u;                           // type=2 ("image") [127:126]
  int32x8 g1;
  g1[0] = (int)(1u << 16);                       // data_size=1 (2 bytes) [17:16]
  g1[1] = (int)(1024u << 16);                    // tensor_dim0=1024 (bits 79:48)
  g1[2] = (int)(1024u << 16);                    // tensor_dim1=1024 (bits 111:80)
  g1[3] = (int)(128u << 16);                     // tile_dim0=128    (bits 127:112)
  g1[4] = 64;                                    // tile_dim1=64     (bits 143:128)
  g1[5] = 1024;                                  // tensor_dim0_stride=1024 (207:160)
  g1[6] = 0;
  g1[7] = 0;
  int32x4 gz = {0, 0, 0, 0};
#if defined(__clang_major__) && (__clang_major__ >= 23)
  int32x8 gz8 = {0, 0, 0, 0, 0, 0, 0, 0};
  __builtin_amdgcn_tensor_load_to_lds(g0, g1, gz, gz, gz8, 0);
#else
  __builtin_amdgcn_tensor_load_to_lds(g0, g1, gz, gz, 0);
#endif
}

// ---------------------------------------------------------------------------
// One-shot f32 -> bf16 conversion (memory bound; removes cvt VALU from GEMMs).
// ---------------------------------------------------------------------------
__global__ __launch_bounds__(256)
void cvt_bf16_k(const float* __restrict__ src, unsigned short* __restrict__ dst, int n) {
  const int i = (blockIdx.x * 256 + threadIdx.x) * 4;
  if (i < n) {
    float4 f = *(const float4*)(src + i);
    ushort4 r;
    r.x = f2bf(f.x); r.y = f2bf(f.y); r.z = f2bf(f.z); r.w = f2bf(f.w);
    *(ushort4*)(dst + i) = r;
  }
}

// ---------------------------------------------------------------------------
// All-bf16 WMMA GEMM: C[M x 1024] = A[M x 1024] * W^T (W row-major [N,K], bf16)
// B operand staged through LDS by the Tensor Data Mover, double buffered in
// 64-col x 128-K chunks (16KB each); wave 0 issues the TDM, all waves consume
// via ds_load fragments. A operand loaded directly (wave-private rows).
// MODE 0: epilogue applies RoPE * oscale, stores bf16 head-major [b,h,s,d] (Q/K)
// MODE 1: epilogue stores bf16 transposed per head [b,h,d,s]               (V)
// MODE 2: epilogue stores f32 [row, col]                                   (out)
// Block: 256 threads = 8 waves; wave tile = 32 rows x 64 cols; BM=256, BN=64.
// ---------------------------------------------------------------------------
template<int MODE>
__global__ __launch_bounds__(256)
void gemm_wmma_k(const unsigned short* __restrict__ A,
                 const unsigned short* __restrict__ W,
                 const int* __restrict__ tokpos, void* __restrict__ Outp,
                 float oscale)
{
  __shared__ __align__(16) unsigned short wtile[2][64 * 128];   // 2 x 16KB

  const int wave = threadIdx.x >> 5;
  const int lane = threadIdx.x & 31;
  const int hf   = lane >> 4;        // lane half (0: lanes 0-15, 1: lanes 16-31)
  const int l15  = lane & 15;
  const int rowTile = blockIdx.y * 256 + wave * 32;
  const int colTile = blockIdx.x * 64;

  const v8f vz = {0.f,0.f,0.f,0.f,0.f,0.f,0.f,0.f};
  v8f acc[2][4] = {{vz, vz, vz, vz}, {vz, vz, vz, vz}};

  const int koffA = hf * 8;          // lane half selects K 0-7/16-23 vs 8-15/24-31
  const unsigned short* arow0 = A + (size_t)(rowTile + l15)      * D_MODEL;
  const unsigned short* arow1 = A + (size_t)(rowTile + 16 + l15) * D_MODEL;
  const unsigned short* wbase = W + (size_t)colTile * D_MODEL;
  const unsigned lds0 = (unsigned)(size_t)(void*)&wtile[0][0];

  // prime buffer 0 with the first K-chunk
  if (wave == 0) {
    tdm_load_w_tile(lds0, wbase);
    __builtin_amdgcn_s_wait_tensorcnt(0);
  }
  __syncthreads();

  for (int kc = 0; kc < D_MODEL; kc += 128) {
    const int bufIdx = (kc >> 7) & 1;
    if ((kc + 128 < D_MODEL) && (wave == 0))     // prefetch next chunk via TDM
      tdm_load_w_tile(lds0 + (bufIdx ^ 1) * (64 * 128 * 2), wbase + kc + 128);

#pragma unroll
    for (int kk = 0; kk < 128; kk += 32) {
      const int kb = kc + kk;
      __builtin_prefetch(arow0 + kb + 64, 0, 0);        // global_prefetch_b8
      BF16Frag a0, a1;
      a0.h[0] = *(const v8bf*)(arow0 + kb + koffA);
      a0.h[1] = *(const v8bf*)(arow0 + kb + 16 + koffA);
      a1.h[0] = *(const v8bf*)(arow1 + kb + koffA);
      a1.h[1] = *(const v8bf*)(arow1 + kb + 16 + koffA);
#pragma unroll
      for (int t = 0; t < 4; ++t) {
        // B fragment from LDS: lane = column, K 0-7/16-23 vs 8-15/24-31 per lane half
        const unsigned short* wl = &wtile[bufIdx][(t * 16 + l15) * 128 + kk + hf * 8];
        BF16Frag b;
        b.h[0] = *(const v8bf*)wl;
        b.h[1] = *(const v8bf*)(wl + 16);
        acc[0][t] = wmma_bf16(a0.v, b.v, acc[0][t]);
        acc[1][t] = wmma_bf16(a1.v, b.v, acc[1][t]);
      }
    }

    if (wave == 0) __builtin_amdgcn_s_wait_tensorcnt(0); // next chunk landed
    __syncthreads();                                     // all done reading cur
  }

  // C layout: VGPR j -> row (hf*8 + j), lane -> col l15
#pragma unroll
  for (int sub = 0; sub < 2; ++sub) {
    const int rbase = rowTile + sub * 16;
#pragma unroll
    for (int t = 0; t < 4; ++t) {
      const int col = colTile + t * 16 + l15;
      if (MODE == 2) {
#pragma unroll
        for (int j = 0; j < 8; ++j) {
          const int row = rbase + hf * 8 + j;
          ((float*)Outp)[(size_t)row * D_MODEL + col] = acc[sub][t][j];
        }
      } else if (MODE == 0) {
        const int h = col >> 6, d = col & 63;
        const float invf = __powf(10000.0f, -(float)(d >> 1) * (1.0f / 32.0f));
#pragma unroll
        for (int j = 0; j < 8; ++j) {
          const int row = rbase + hf * 8 + j;
          float v  = acc[sub][t][j];
          float pr = __shfl_xor(v, 1);            // partner element (col^1, same row)
          const int b = row >> 11, s = row & (SEQ - 1);
          const float ang = (float)tokpos[s] * invf;
          float sn, cs; __sincosf(ang, &sn, &cs);
          const float r = ((d & 1) ? (v * cs + pr * sn) : (v * cs - pr * sn)) * oscale;
          ((unsigned short*)Outp)[((((size_t)b * NUM_HEADS + h) * SEQ + s) << 6) + d] = f2bf(r);
        }
      } else { // MODE 1: V transposed [b,h,d,s]
        const int h = col >> 6, d = col & 63;
#pragma unroll
        for (int j = 0; j < 8; ++j) {
          const int row = rbase + hf * 8 + j;
          const int b = row >> 11, s = row & (SEQ - 1);
          ((unsigned short*)Outp)[(((size_t)b * NUM_HEADS + h) * D_K + d) * SEQ + s] =
              f2bf(acc[sub][t][j]);
        }
      }
    }
  }
}

// ---------------------------------------------------------------------------
// Flash attention, causal. Grid: (bh=32, qblk=32). Block: 128 = 4 waves.
// Wave handles a 16-row Q tile; streams 64-key blocks (BK=64).
// Q (pre-scaled by 1/sqrt(dk)), K: bf16 [b,h,s,64]. Vt: bf16 [b,h,64,s].
// Softmax denominator accumulated via WMMA against an all-ones B fragment.
// Causal mask applied only on the final (diagonal) key block.
// ---------------------------------------------------------------------------
__global__ __launch_bounds__(128)
void attn_k(const unsigned short* __restrict__ Q,
            const unsigned short* __restrict__ K,
            const unsigned short* __restrict__ Vt,
            unsigned short* __restrict__ AO)
{
  __shared__ __align__(16) unsigned short pbuf[4][16 * 64];  // per-wave P bounce (8KB)

  const int wave = threadIdx.x >> 5;
  const int lane = threadIdx.x & 31;
  const int hf   = lane >> 4;
  const int l15  = lane & 15;
  const int bh   = blockIdx.x;                 // b*16 + h
  const int qblk = blockIdx.y;
  const int qbase = qblk * 64 + wave * 16;

  const unsigned short* Qp = Q  + (size_t)bh * SEQ * D_K;
  const unsigned short* Kp = K  + (size_t)bh * SEQ * D_K;
  const unsigned short* Vp = Vt + (size_t)bh * D_K * SEQ;

  // Q A-fragments for d-chunks 0..31 and 32..63
  BF16Frag aq[2];
  {
    const int qrow  = qbase + l15;
    const int koffA = hf * 8;
#pragma unroll
    for (int dc = 0; dc < 2; ++dc) {
      aq[dc].h[0] = *(const v8bf*)(Qp + (size_t)qrow * D_K + dc * 32 + koffA);
      aq[dc].h[1] = *(const v8bf*)(Qp + (size_t)qrow * D_K + dc * 32 + 16 + koffA);
    }
  }

  // all-ones B fragment (32-bit splat): P @ ones -> row softmax denominator
  BF16Frag ones;
#pragma unroll
  for (int i = 0; i < 8; ++i) ones.w[i] = 0x3F803F80u;   // two bf16 1.0 per dword

  const v8f vz = {0.f,0.f,0.f,0.f,0.f,0.f,0.f,0.f};
  v8f o[4] = {vz, vz, vz, vz};
  v8f lsum = vz;                                      // denominator accumulator
  float m[8];
#pragma unroll
  for (int j = 0; j < 8; ++j) m[j] = -1e30f;

  const int nkb = qblk + 1;          // 64-key blocks; all active for every wave
  for (int kb = 0; kb < nkb; ++kb) {
    const int k0 = kb * 64;
    const bool diag = (kb == nkb - 1);   // wave-uniform: mask only diagonal block

    // S = Q * K^T  (16 x 64 block: 4 N-tiles x 2 K-steps = 8 WMMAs)
    v8f s[4] = {vz, vz, vz, vz};
#pragma unroll
    for (int dc = 0; dc < 2; ++dc) {
#pragma unroll
      for (int t = 0; t < 4; ++t) {
        BF16Frag b;
        const unsigned short* p = Kp + (size_t)(k0 + t * 16 + l15) * D_K + dc * 32 + hf * 8;
        b.h[0] = *(const v8bf*)p; b.h[1] = *(const v8bf*)(p + 16);
        s[t] = wmma_bf16(aq[dc].v, b.v, s[t]);
      }
    }

    // online softmax update (row lives across 16 lanes of one half)
#pragma unroll
    for (int j = 0; j < 8; ++j) {
      const int row = qbase + hf * 8 + j;
      float v0 = s[0][j], v1 = s[1][j], v2 = s[2][j], v3 = s[3][j];
      if (diag) {                       // causal mask, diagonal block only
        if (k0 + 0  + l15 > row) v0 = -1e9f;
        if (k0 + 16 + l15 > row) v1 = -1e9f;
        if (k0 + 32 + l15 > row) v2 = -1e9f;
        if (k0 + 48 + l15 > row) v3 = -1e9f;
      }
      float mj = fmaxf(fmaxf(v0, v1), fmaxf(v2, v3));
#pragma unroll
      for (int off = 1; off < 16; off <<= 1) mj = fmaxf(mj, __shfl_xor(mj, off));
      const float mn   = fmaxf(m[j], mj);
      const float corr = __expf(m[j] - mn);
      m[j] = mn;
      o[0][j] *= corr; o[1][j] *= corr; o[2][j] *= corr; o[3][j] *= corr;
      lsum[j] *= corr;
      unsigned short* pr = &pbuf[wave][(hf * 8 + j) * 64 + l15];
      pr[0]  = f2bf(__expf(v0 - mn));
      pr[16] = f2bf(__expf(v1 - mn));
      pr[32] = f2bf(__expf(v2 - mn));
      pr[48] = f2bf(__expf(v3 - mn));
    }
    __syncthreads();   // uniform; orders DS writes vs cross-lane DS reads

    // re-read P (16x64) in A-matrix layout: two K=32 fragments
    BF16Frag pa0, pa1;
    pa0.h[0] = *(const v8bf*)&pbuf[wave][l15 * 64 + hf * 8];
    pa0.h[1] = *(const v8bf*)&pbuf[wave][l15 * 64 + 16 + hf * 8];
    pa1.h[0] = *(const v8bf*)&pbuf[wave][l15 * 64 + 32 + hf * 8];
    pa1.h[1] = *(const v8bf*)&pbuf[wave][l15 * 64 + 48 + hf * 8];

    // O += P * V ; denominator += P * ones
#pragma unroll
    for (int t = 0; t < 4; ++t) {
      BF16Frag bv0, bv1;
      const unsigned short* vb = Vp + (size_t)(t * 16 + l15) * SEQ + k0 + hf * 8;
      bv0.h[0] = *(const v8bf*)vb;        bv0.h[1] = *(const v8bf*)(vb + 16);
      bv1.h[0] = *(const v8bf*)(vb + 32); bv1.h[1] = *(const v8bf*)(vb + 48);
      o[t] = wmma_bf16(pa0.v, bv0.v, o[t]);
      o[t] = wmma_bf16(pa1.v, bv1.v, o[t]);
    }
    lsum = wmma_bf16(pa0.v, ones.v, lsum);
    lsum = wmma_bf16(pa1.v, ones.v, lsum);
  }

  // normalize and store bf16 to AO[b*s, h*64+d]
  const int b = bh >> 4, h = bh & 15;
  float inv[8];
#pragma unroll
  for (int j = 0; j < 8; ++j) inv[j] = 1.0f / lsum[j];
#pragma unroll
  for (int t = 0; t < 4; ++t) {
    const int col = h * 64 + t * 16 + l15;
#pragma unroll
    for (int j = 0; j < 8; ++j) {
      const int row = b * SEQ + qbase + hf * 8 + j;
      AO[(size_t)row * D_MODEL + col] = f2bf(o[t][j] * inv[j]);
    }
  }
}

// ---------------------------------------------------------------------------
extern "C" void kernel_launch(void* const* d_in, const int* in_sizes, int n_in,
                              void* d_out, int out_size, void* d_ws, size_t ws_size,
                              hipStream_t stream) {
  const float* x      = (const float*)d_in[0];
  const int*   tokpos = (const int*)d_in[1];
  const float* Wq     = (const float*)d_in[2];
  const float* Wk     = (const float*)d_in[3];
  const float* Wv     = (const float*)d_in[4];
  const float* Wo     = (const float*)d_in[5];

  const size_t me = (size_t)M_TOTAL * D_MODEL;   // 4096*1024
  const size_t we = (size_t)D_MODEL * D_MODEL;   // 1024*1024
  unsigned short* Qh  = (unsigned short*)d_ws;
  unsigned short* Kh  = Qh  + me;
  unsigned short* Vt  = Kh  + me;
  unsigned short* AO  = Vt  + me;
  unsigned short* Xb  = AO  + me;
  unsigned short* Wqb = Xb  + me;
  unsigned short* Wkb = Wqb + we;
  unsigned short* Wvb = Wkb + we;
  unsigned short* Wob = Wvb + we;

  // one-shot f32 -> bf16 conversions
  cvt_bf16_k<<<(int)(me / 4 / 256), 256, 0, stream>>>(x,  Xb,  (int)me);
  cvt_bf16_k<<<(int)(we / 4 / 256), 256, 0, stream>>>(Wq, Wqb, (int)we);
  cvt_bf16_k<<<(int)(we / 4 / 256), 256, 0, stream>>>(Wk, Wkb, (int)we);
  cvt_bf16_k<<<(int)(we / 4 / 256), 256, 0, stream>>>(Wv, Wvb, (int)we);
  cvt_bf16_k<<<(int)(we / 4 / 256), 256, 0, stream>>>(Wo, Wob, (int)we);

  dim3 gProj(D_MODEL / 64, M_TOTAL / 256);   // (16, 16)
  // Q pre-scaled by 1/sqrt(dk) = 0.125 so attention needs no score scaling
  gemm_wmma_k<0><<<gProj, 256, 0, stream>>>(Xb, Wqb, tokpos, (void*)Qh, 0.125f);
  gemm_wmma_k<0><<<gProj, 256, 0, stream>>>(Xb, Wkb, tokpos, (void*)Kh, 1.0f);
  gemm_wmma_k<1><<<gProj, 256, 0, stream>>>(Xb, Wvb, tokpos, (void*)Vt, 1.0f);

  attn_k<<<dim3(BATCH * NUM_HEADS, SEQ / 64), 128, 0, stream>>>(Qh, Kh, Vt, AO);

  gemm_wmma_k<2><<<gProj, 256, 0, stream>>>(AO, Wob, tokpos, d_out, 1.0f);
}
